// MHSA_13331578487365
// MI455X (gfx1250) — compile-verified, matrix operand order checked by reading, not run on hardware
//
#include <hip/hip_runtime.h>
#include <hip/hip_bf16.h>

// MHSA: B=2, N=2048, C=1024, H=16, HS=64, SCALE=1/8.
// Pipeline:
//   (0) one-shot fp32->bf16 conversion of x, w_qkv, w_out (hardware cvt)
//   (1) QKV GEMM (pure bf16 loads + WMMA) w/ stride-3 interleave decode
//   (2) flash attention (streaming softmax, P via LDS, V^T layout)
//   (3) out-projection GEMM + bias -> fp32 d_out
// All matrix math: v_wmma_f32_16x16x32_bf16 (wave32).
// Workspace requirement: 48 MB.

#define Bb   2
#define Nn   2048
#define Cc   1024
#define Hh   16
#define HS   64
#define SCALE 0.125f

typedef __attribute__((ext_vector_type(16))) __bf16 v16bf;
typedef __attribute__((ext_vector_type(8)))  __bf16 v8bf;
typedef __attribute__((ext_vector_type(8)))  float  v8f;
typedef __attribute__((ext_vector_type(8)))  unsigned short us8;

union BFrag  { v16bf v; unsigned short u[16]; us8 h[2]; };
union BPack8 { v8bf b; us8 u; };

__device__ __forceinline__ unsigned short f2bf(float f) {
    union { __bf16 b; unsigned short u; } c;
    c.b = static_cast<__bf16>(f);              // hardware v_cvt (RNE)
    return c.u;
}

// Fragment element mapping (A 16x32 and B 32x16 share it):
//  lane L (hi=L/16, lo=L%16): row/col = lo; element e:
//  k = (e/8)*16 + hi*8 + (e%8)  ->  p points at rowstart + k0 + hi*8,
//  elements are p[0..7] and p[16..23] (two contiguous 16B groups).
__device__ __forceinline__ v16bf frag_from_bf16(const unsigned short* p) {
    BFrag fr;
    fr.h[0] = *(const us8*)(p);
    fr.h[1] = *(const us8*)(p + 16);
    return fr.v;
}

__device__ __forceinline__ v8f wmma_bf16(v16bf a, v16bf b, v8f c) {
    return __builtin_amdgcn_wmma_f32_16x16x32_bf16(
        false, a, false, b, (short)0, c, false, false);
}

// ---------------------------------------------------------------------------
// Kernel 0: elementwise fp32 -> bf16 (8 elements/thread, vector cvt+b128 io)
// ---------------------------------------------------------------------------
__global__ __launch_bounds__(256)
void cvt_f32_bf16(const float* __restrict__ src, unsigned short* __restrict__ dst,
                  int n8) {
    const int i = blockIdx.x * 256 + threadIdx.x;
    if (i < n8) {
        v8f a = *(const v8f*)(src + (size_t)i * 8);
        BPack8 p;
        p.b = __builtin_convertvector(a, v8bf);
        *(us8*)(dst + (size_t)i * 8) = p.u;
    }
}

// ---------------------------------------------------------------------------
// Kernel 1: qkv = x @ w_qkv^T, de-interleave (d%3 -> q/k/v, j=d/3=h*64+hs).
// Wave tile: 32 rows x 64 cols. Grid: (4096/32)*(3072/64)=6144 waves.
// Inner loop: 6 b128-pair loads + 8 WMMA, zero conversions.
// ---------------------------------------------------------------------------
__global__ __launch_bounds__(128)
void qkv_kernel(const unsigned short* __restrict__ xbf,
                const unsigned short* __restrict__ wbf,
                unsigned short* __restrict__ qo, unsigned short* __restrict__ ko,
                unsigned short* __restrict__ vo) {
    const int lane = threadIdx.x & 31;
    const int lo = lane & 15, hi = lane >> 4;
    const int wave = blockIdx.x * 4 + (threadIdx.x >> 5);
    const int DB = (3 * Cc) / 64;               // 48
    const int mb = wave / DB, db = wave % DB;
    const int m0 = mb * 32, d0 = db * 64;

    v8f acc[2][4] = {};
    for (int k0 = 0; k0 < Cc; k0 += 32) {
        v16bf a0 = frag_from_bf16(xbf + (size_t)(m0 + lo) * Cc + k0 + hi * 8);
        v16bf a1 = frag_from_bf16(xbf + (size_t)(m0 + 16 + lo) * Cc + k0 + hi * 8);
#pragma unroll
        for (int c = 0; c < 4; ++c) {
            v16bf bf = frag_from_bf16(wbf + (size_t)(d0 + c * 16 + lo) * Cc + k0 + hi * 8);
            acc[0][c] = wmma_bf16(a0, bf, acc[0][c]);
            acc[1][c] = wmma_bf16(a1, bf, acc[1][c]);
        }
    }
#pragma unroll
    for (int mr = 0; mr < 2; ++mr) {
#pragma unroll
        for (int c = 0; c < 4; ++c) {
            const int d = d0 + c * 16 + lo;
            const int which = d % 3, j = d / 3;
            const int h = j >> 6, hs = j & 63;
#pragma unroll
            for (int r = 0; r < 8; ++r) {
                const int m = m0 + mr * 16 + r + 8 * hi;
                const int b = m >> 11, n = m & (Nn - 1);
                const float val = acc[mr][c][r];
                const size_t qidx = ((size_t)(b * Hh + h) * Nn + n) * HS + hs;
                if (which == 0)      qo[qidx] = f2bf(val * SCALE);
                else if (which == 1) ko[qidx] = f2bf(val);
                else vo[((size_t)(b * Hh + h) * HS + hs) * Nn + n] = f2bf(val);
            }
        }
    }
}

// ---------------------------------------------------------------------------
// Kernel 2: flash attention. One wave per (b,h, 16-query block) = 4096 waves.
// Key blocks of 32: 4 WMMA for S, online softmax, P via LDS, 4 WMMA for O+=PV.
// ---------------------------------------------------------------------------
__global__ __launch_bounds__(128)
void attn_kernel(const unsigned short* __restrict__ q_,
                 const unsigned short* __restrict__ k_,
                 const unsigned short* __restrict__ vT,
                 unsigned short* __restrict__ o_) {
    __shared__ unsigned short ldsP[4][16 * 32];   // 1 KB per wave
    const int lane = threadIdx.x & 31;
    const int lo = lane & 15, hi = lane >> 4;
    const int wib = threadIdx.x >> 5;
    const int wave = blockIdx.x * 4 + wib;
    const int bh = wave >> 7;                     // b*16 + h
    const int qb = wave & 127;
    unsigned short* pl = ldsP[wib];

    const unsigned short* qbase = q_ + ((size_t)bh * Nn + qb * 16) * HS;
    const v16bf aq0 = frag_from_bf16(qbase + lo * HS + 0  + hi * 8);
    const v16bf aq1 = frag_from_bf16(qbase + lo * HS + 32 + hi * 8);
    const unsigned short* kbase = k_ + (size_t)bh * Nn * HS;
    const unsigned short* vbase = vT + (size_t)bh * HS * Nn;

    float mrow[8], lrow[8];
#pragma unroll
    for (int r = 0; r < 8; ++r) { mrow[r] = -INFINITY; lrow[r] = 0.f; }
    v8f acc[4] = {};

    for (int kb = 0; kb < Nn; kb += 32) {
        // S = (SCALE*Q) @ K^T : two 16x16 tiles covering 32 keys
        v8f s0 = {}, s1 = {};
        {
            const unsigned short* kr0 = kbase + (size_t)(kb + lo) * HS;
            s0 = wmma_bf16(aq0, frag_from_bf16(kr0 + 0  + hi * 8), s0);
            s0 = wmma_bf16(aq1, frag_from_bf16(kr0 + 32 + hi * 8), s0);
            const unsigned short* kr1 = kbase + (size_t)(kb + 16 + lo) * HS;
            s1 = wmma_bf16(aq0, frag_from_bf16(kr1 + 0  + hi * 8), s1);
            s1 = wmma_bf16(aq1, frag_from_bf16(kr1 + 32 + hi * 8), s1);
        }
        // online softmax across the 32 keys (row m = r + 8*hi lives in 16 lanes)
        float corr[8];
#pragma unroll
        for (int r = 0; r < 8; ++r) {
            float mx = fmaxf(s0[r], s1[r]);
#pragma unroll
            for (int ms = 1; ms <= 8; ms <<= 1) mx = fmaxf(mx, __shfl_xor(mx, ms, 32));
            const float mnew = fmaxf(mrow[r], mx);
            const float p0 = __expf(s0[r] - mnew);
            const float p1 = __expf(s1[r] - mnew);
            float sum = p0 + p1;
#pragma unroll
            for (int ms = 1; ms <= 8; ms <<= 1) sum += __shfl_xor(sum, ms, 32);
            const float c = __expf(mrow[r] - mnew);
            lrow[r] = lrow[r] * c + sum;
            mrow[r] = mnew;
            corr[r] = c;
            s0[r] = p0; s1[r] = p1;
        }
        // rescale running O accumulator
#pragma unroll
        for (int c = 0; c < 4; ++c)
#pragma unroll
            for (int r = 0; r < 8; ++r) acc[c][r] *= corr[r];
        // P (16x32) through LDS into A-fragment layout
#pragma unroll
        for (int r = 0; r < 8; ++r) {
            pl[(r + 8 * hi) * 32 + lo]      = f2bf(s0[r]);
            pl[(r + 8 * hi) * 32 + 16 + lo] = f2bf(s1[r]);
        }
        asm volatile("s_wait_dscnt 0x0" ::: "memory");
        const v16bf pa = frag_from_bf16(pl + lo * 32 + hi * 8);
        // O += P @ V   (V^T layout: contiguous along keys)
#pragma unroll
        for (int c = 0; c < 4; ++c) {
            const v16bf vb = frag_from_bf16(vbase + (size_t)(c * 16 + lo) * Nn + kb + hi * 8);
            acc[c] = wmma_bf16(pa, vb, acc[c]);
        }
    }
    // normalize and store attn in (b, n, h*64+d) = (B,N,C) bf16
    const int b = bh >> 4, h = bh & 15;
    float inv[8];
#pragma unroll
    for (int r = 0; r < 8; ++r) inv[r] = 1.f / lrow[r];
#pragma unroll
    for (int c = 0; c < 4; ++c)
#pragma unroll
        for (int r = 0; r < 8; ++r) {
            const int n = qb * 16 + r + 8 * hi;
            o_[((size_t)(b * Nn + n)) * Cc + h * HS + c * 16 + lo] =
                f2bf(acc[c][r] * inv[r]);
        }
}

// ---------------------------------------------------------------------------
// Kernel 3: out = attn @ w_out^T + b_out (fp32 out). 32x64 wave tiles.
// ---------------------------------------------------------------------------
__global__ __launch_bounds__(128)
void proj_kernel(const unsigned short* __restrict__ attn,
                 const unsigned short* __restrict__ wbf,
                 const float* __restrict__ bias,
                 float* __restrict__ out) {
    const int lane = threadIdx.x & 31;
    const int lo = lane & 15, hi = lane >> 4;
    const int wave = blockIdx.x * 4 + (threadIdx.x >> 5);
    const int DB = Cc / 64;                     // 16
    const int mb = wave / DB, db = wave % DB;
    const int m0 = mb * 32, d0 = db * 64;

    v8f acc[2][4] = {};
    for (int k0 = 0; k0 < Cc; k0 += 32) {
        v16bf a0 = frag_from_bf16(attn + (size_t)(m0 + lo) * Cc + k0 + hi * 8);
        v16bf a1 = frag_from_bf16(attn + (size_t)(m0 + 16 + lo) * Cc + k0 + hi * 8);
#pragma unroll
        for (int c = 0; c < 4; ++c) {
            v16bf bf = frag_from_bf16(wbf + (size_t)(d0 + c * 16 + lo) * Cc + k0 + hi * 8);
            acc[0][c] = wmma_bf16(a0, bf, acc[0][c]);
            acc[1][c] = wmma_bf16(a1, bf, acc[1][c]);
        }
    }
#pragma unroll
    for (int c = 0; c < 4; ++c) {
        const int d = d0 + c * 16 + lo;
        const float bia = bias[d];
#pragma unroll
        for (int mr = 0; mr < 2; ++mr)
#pragma unroll
            for (int r = 0; r < 8; ++r) {
                const int m = m0 + mr * 16 + r + 8 * hi;
                out[(size_t)m * Cc + d] = acc[mr][c][r] + bia;
            }
    }
}

// ---------------------------------------------------------------------------
extern "C" void kernel_launch(void* const* d_in, const int* in_sizes, int n_in,
                              void* d_out, int out_size, void* d_ws, size_t ws_size,
                              hipStream_t stream) {
    (void)in_sizes; (void)n_in; (void)out_size; (void)ws_size;
    const float* x     = (const float*)d_in[0];
    const float* w_qkv = (const float*)d_in[1];
    const float* w_out = (const float*)d_in[2];
    const float* b_out = (const float*)d_in[3];

    const size_t NX  = (size_t)Bb * Nn * Cc;          // 4 Mi
    const size_t NWQ = (size_t)3 * Cc * Cc;           // 3 Mi
    const size_t NWO = (size_t)Cc * Cc;               // 1 Mi
    const size_t NHD = (size_t)Bb * Hh * Nn * HS;     // 4 Mi

    unsigned short* xbf  = (unsigned short*)d_ws;     // 8 MB
    unsigned short* wqb  = xbf + NX;                  // 6 MB
    unsigned short* wob  = wqb + NWQ;                 // 2 MB
    unsigned short* qws  = wob + NWO;                 // 8 MB
    unsigned short* kws  = qws + NHD;                 // 8 MB
    unsigned short* vws  = kws + NHD;                 // 8 MB
    unsigned short* aws  = vws + NHD;                 // 8 MB  (total 48 MB)

    cvt_f32_bf16<<<(int)(NX  / 8 / 256), 256, 0, stream>>>(x,     xbf, (int)(NX  / 8));
    cvt_f32_bf16<<<(int)(NWQ / 8 / 256), 256, 0, stream>>>(w_qkv, wqb, (int)(NWQ / 8));
    cvt_f32_bf16<<<(int)(NWO / 8 / 256), 256, 0, stream>>>(w_out, wob, (int)(NWO / 8));

    // 6144 waves / 4 per block
    qkv_kernel<<<1536, 128, 0, stream>>>(xbf, wqb, qws, kws, vws);
    // 4096 waves / 4 per block
    attn_kernel<<<1024, 128, 0, stream>>>(qws, kws, vws, aws);
    // 2048 waves / 4 per block
    proj_kernel<<<512, 128, 0, stream>>>(aws, wob, b_out, (float*)d_out);
}
